// GlobalSubSampleAttn_81449759801421
// MI455X (gfx1250) — matrix operand order repaired and checked by reading, not compiled, not run on hardware
//
#include <hip/hip_runtime.h>
#include <hip/hip_bf16.h>
#include <math.h>

// ---------------------------------------------------------------------------
// GlobalSubSampleAttn for MI455X (gfx1250, wave32, WMMA)
// All heavy GEMMs run through v_wmma_f32_16x16x32_bf16 (fp32 accumulate).
// Weights kept in their native [N_out, K] layout so both A and B tiles stage
// into LDS with aligned b128 copies (async-to-LDS when the builtin exists).
// ---------------------------------------------------------------------------

typedef __bf16 bf16_t;
typedef __attribute__((ext_vector_type(16))) __bf16 v16bf;
typedef __attribute__((ext_vector_type(8)))  __bf16 v8bf;
typedef __attribute__((ext_vector_type(8)))  float  v8f;
typedef int v4i_vec __attribute__((vector_size(16)));   // matches builtin param type

#define AS_GLOBAL __attribute__((address_space(1)))
#define AS_LDS    __attribute__((address_space(3)))

#if defined(__has_builtin)
# if __has_builtin(__builtin_amdgcn_global_load_async_to_lds_b128) && \
     __has_builtin(__builtin_amdgcn_s_wait_asynccnt)
#  define USE_ASYNC_COPY 1
# endif
#endif
#ifndef USE_ASYNC_COPY
# define USE_ASYNC_COPY 0
#endif

static __device__ __forceinline__ bf16_t f2bf(float f) {
  // round-to-nearest-even fp32 -> bf16
  unsigned u = __builtin_bit_cast(unsigned, f);
  unsigned r = u + 0x7FFFu + ((u >> 16) & 1u);
  unsigned short h = (unsigned short)(r >> 16);
  return __builtin_bit_cast(bf16_t, h);
}

// ---------------- pack / layout kernels ------------------------------------

// x [32768,256] f32 -> bf16 into left half of concat [32768,512]
__global__ __launch_bounds__(256) void pack_x_concat(const float* __restrict__ x,
                                                     bf16_t* __restrict__ cat) {
  int i = blockIdx.x * 256 + threadIdx.x;       // 0 .. 8388607
  int row = i >> 8, col = i & 255;
  cat[(size_t)row * 512 + col] = f2bf(x[i]);
}

// gather 4x4 patches: A_conv[(b*256+s), p*256+ch] = x[b, pixel(s,p), ch]
__global__ __launch_bounds__(256) void pack_conv_a(const float* __restrict__ x,
                                                   bf16_t* __restrict__ Ac) {
  int i = blockIdx.x * 256 + threadIdx.x;       // 0 .. 2048*4096-1
  int row = i >> 12;                            // b*256+s
  int rem = i & 4095;
  int p = rem >> 8, ch = rem & 255;
  int b = row >> 8, s = row & 255;
  int sy = s >> 4, sx = s & 15;
  int py = p >> 2, px = p & 3;
  int pix = (sy * 4 + py) * 64 + (sx * 4 + px);
  Ac[i] = f2bf(x[((size_t)b * 4096 + pix) * 256 + ch]);
}

// sr_w OIHW [256,256,4,4] -> Wc[o, k] with k = (kh*4+kw)*256 + ch  ([N,K] layout)
__global__ __launch_bounds__(256) void pack_conv_w(const float* __restrict__ w,
                                                   bf16_t* __restrict__ Wc) {
  int i = blockIdx.x * 256 + threadIdx.x;       // 0 .. 256*4096-1
  int o = i >> 12, k = i & 4095;
  int p = k >> 8, ch = k & 255;
  int kh = p >> 2, kw = p & 3;
  Wc[i] = f2bf(w[(((size_t)o * 256 + ch) * 4 + kh) * 4 + kw]);
}

// straight fp32 -> bf16 convert (weights stay in native [N,K] layout)
__global__ __launch_bounds__(256) void pack_cvt(const float* __restrict__ w,
                                                bf16_t* __restrict__ wt, int n) {
  int i = blockIdx.x * 256 + threadIdx.x;
  if (i < n) wt[i] = f2bf(w[i]);
}

// ---------------- WMMA GEMM ------------------------------------------------
// C[M,N] = A[M,K](bf16, row-major lda) * B[N,K](bf16, row-major ldb)^T
// Tile 128x128x32, 256 threads = 8 waves (2 x 4), 4x2 wmma tiles per wave.
// M,N multiples of 128; K multiple of 32 (all call sites satisfy this).

enum { EPI_NONE = 0, EPI_BIAS = 1, EPI_ELU1 = 2, EPI_RELU = 3 };

template <int EPI>
__global__ __launch_bounds__(256)
void gemm_bf16_wmma(const bf16_t* __restrict__ A, int lda,
                    const bf16_t* __restrict__ Bw, int ldb,
                    float* __restrict__ Cf, bf16_t* __restrict__ Cb, int ldc,
                    int K, const float* __restrict__ bias) {
  constexpr int BK = 32, LDSS = 48;             // +16 bf16 pad, rows stay 16B aligned
  __shared__ bf16_t sA[128][LDSS];              // sA[m][k]
  __shared__ bf16_t sB[128][LDSS];              // sB[n][k] (weights already K-major)

  const int t = threadIdx.x;
  const int m0 = blockIdx.x * 128;
  const int n0 = blockIdx.y * 128;
  const int wave = t >> 5, lane = t & 31;
  const int wm = (wave >> 2) * 64;              // 0 or 64
  const int wn = (wave & 3) * 32;               // 0,32,64,96
  const int ml = lane & 15, kh = lane >> 4;

  v8f acc[4][2];
#pragma unroll
  for (int i = 0; i < 4; ++i)
#pragma unroll
    for (int j = 0; j < 2; ++j)
#pragma unroll
      for (int e = 0; e < 8; ++e) acc[i][j][e] = 0.f;

  for (int kt = 0; kt < K; kt += BK) {
    if (kt + BK < K) {                          // L2 prefetch of next tiles
      int vec = t;
      __builtin_prefetch(A + (size_t)(m0 + (vec >> 2)) * lda + kt + BK + ((vec & 3) << 3), 0, 1);
      __builtin_prefetch(Bw + (size_t)(n0 + (vec >> 2)) * ldb + kt + BK + ((vec & 3) << 3), 0, 1);
    }
#if USE_ASYNC_COPY
    // one-instruction global->LDS copies, tracked by ASYNCcnt
#pragma unroll
    for (int i = 0; i < 2; ++i) {
      int vec = t + i * 256;                    // 512 vectors of 8 bf16 per tile
      int row = vec >> 2;
      int kc = (vec & 3) << 3;
      __builtin_amdgcn_global_load_async_to_lds_b128(
          (AS_GLOBAL v4i_vec*)(A + (size_t)(m0 + row) * lda + (kt + kc)),
          (AS_LDS v4i_vec*)&sA[row][kc], 0, 0);
      __builtin_amdgcn_global_load_async_to_lds_b128(
          (AS_GLOBAL v4i_vec*)(Bw + (size_t)(n0 + row) * ldb + (kt + kc)),
          (AS_LDS v4i_vec*)&sB[row][kc], 0, 0);
    }
    __builtin_amdgcn_s_wait_asynccnt(0);
    __syncthreads();
#else
#pragma unroll
    for (int i = 0; i < 2; ++i) {
      int vec = t + i * 256;
      int row = vec >> 2;
      int kc = (vec & 3) << 3;
      *reinterpret_cast<v8bf*>(&sA[row][kc]) =
          *reinterpret_cast<const v8bf*>(A + (size_t)(m0 + row) * lda + (kt + kc));
      *reinterpret_cast<v8bf*>(&sB[row][kc]) =
          *reinterpret_cast<const v8bf*>(Bw + (size_t)(n0 + row) * ldb + (kt + kc));
    }
    __syncthreads();
#endif

    // A frag (ISA 16-bit 16x32 layout): lanes 0-15 K{0..7,16..23}, lanes 16-31 K{8..15,24..31}
    v16bf af[4], bfr[2];
#pragma unroll
    for (int im = 0; im < 4; ++im) {
      int r = wm + im * 16 + ml;
      v8bf lo = *reinterpret_cast<const v8bf*>(&sA[r][kh * 8]);
      v8bf hi = *reinterpret_cast<const v8bf*>(&sA[r][16 + kh * 8]);
#pragma unroll
      for (int e = 0; e < 8; ++e) { af[im][e] = lo[e]; af[im][e + 8] = hi[e]; }
    }
    // B frag: lane holds column (lane&15), K{0..7,16..23} for lanes 0-15, K{8..15,24..31} for lanes 16-31
#pragma unroll
    for (int in_ = 0; in_ < 2; ++in_) {
      int c = wn + in_ * 16 + ml;
      v8bf lo = *reinterpret_cast<const v8bf*>(&sB[c][kh * 8]);
      v8bf hi = *reinterpret_cast<const v8bf*>(&sB[c][16 + kh * 8]);
#pragma unroll
      for (int e = 0; e < 8; ++e) { bfr[in_][e] = lo[e]; bfr[in_][e + 8] = hi[e]; }
    }
#pragma unroll
    for (int im = 0; im < 4; ++im)
#pragma unroll
      for (int in_ = 0; in_ < 2; ++in_)
        acc[im][in_] = __builtin_amdgcn_wmma_f32_16x16x32_bf16(
            false, af[im], false, bfr[in_], (short)0, acc[im][in_], false, false);
    __syncthreads();
  }

  // epilogue; C/D layout: lanes 0-15 -> N=lane, M=vgpr; lanes 16-31 -> N=lane-16, M=8+vgpr
  const int colt = lane & 15;
  const int rb = (lane >> 4) * 8;
#pragma unroll
  for (int im = 0; im < 4; ++im)
#pragma unroll
    for (int in_ = 0; in_ < 2; ++in_) {
      int gc = n0 + wn + in_ * 16 + colt;
      float bv = (EPI == EPI_BIAS) ? bias[gc] : 0.f;
#pragma unroll
      for (int r = 0; r < 8; ++r) {
        int gr = m0 + wm + im * 16 + rb + r;
        float v = acc[im][in_][r];
        if (EPI == EPI_BIAS) v += bv;
        else if (EPI == EPI_ELU1) v = v > 0.f ? v + 1.f : __expf(v);  // elu(x)+1
        else if (EPI == EPI_RELU) v = v > 0.f ? v : 0.f;
        if (Cf) Cf[(size_t)gr * ldc + gc] = v;
        if (Cb) Cb[(size_t)gr * ldc + gc] = f2bf(v);
      }
    }
}

// ---------------- LayerNorm over 256 columns -------------------------------
__global__ __launch_bounds__(256)
void ln256(const float* __restrict__ in, const float* __restrict__ g,
           const float* __restrict__ be, float* __restrict__ outf,
           bf16_t* __restrict__ outb, int ldb, int coloff,
           const float* __restrict__ res) {
  __shared__ float red[256];
  int row = blockIdx.x, t = threadIdx.x;
  float v = in[(size_t)row * 256 + t];
  red[t] = v; __syncthreads();
  for (int s = 128; s > 0; s >>= 1) { if (t < s) red[t] += red[t + s]; __syncthreads(); }
  float mean = red[0] * (1.f / 256.f);
  __syncthreads();
  float d = v - mean;
  red[t] = d * d; __syncthreads();
  for (int s = 128; s > 0; s >>= 1) { if (t < s) red[t] += red[t + s]; __syncthreads(); }
  float var = red[0] * (1.f / 256.f);
  float y = d * rsqrtf(var + 1e-5f) * g[t] + be[t];
  if (res) y += res[(size_t)row * 256 + t];
  if (outf) outf[(size_t)row * 256 + t] = y;
  if (outb) outb[(size_t)row * ldb + coloff + t] = f2bf(y);
}

// ---------------- linear-attention reductions ------------------------------
// KV[b,h,d,v] = sum_s K[b,s,h,d]*V[b,s,h,v]/S ;  Ksum[b,h,d] = sum_s K
__global__ __launch_bounds__(256)
void kv_reduce(const float* __restrict__ Kf, const float* __restrict__ Vf,
               float* __restrict__ KV, float* __restrict__ Ksum) {
  int bh = blockIdx.x;                          // b*8+h
  int b = bh >> 3, h = bh & 7;
  __shared__ float sk[32], sv[32];
  int t = threadIdx.x;
  int d = t >> 3;
  int v0 = (t & 7) * 4;
  float acc[4] = {0.f, 0.f, 0.f, 0.f};
  float ks = 0.f;
  for (int s = 0; s < 256; ++s) {
    size_t base = ((size_t)b * 256 + s) * 256 + h * 32;
    if (t < 32) sk[t] = Kf[base + t];
    else if (t < 64) sv[t - 32] = Vf[base + (t - 32)];
    __syncthreads();
    float kd = sk[d];
#pragma unroll
    for (int j = 0; j < 4; ++j) acc[j] += kd * sv[v0 + j];
    if (t < 32) ks += sk[t];
    __syncthreads();
  }
#pragma unroll
  for (int j = 0; j < 4; ++j) KV[(size_t)bh * 1024 + d * 32 + v0 + j] = acc[j] * (1.f / 256.f);
  if (t < 32) Ksum[bh * 32 + t] = ks;
}

// msg[b,l,h,v] = (sum_d Q*KV) / (sum_d Q*Ksum + eps) * S, packed bf16
__global__ __launch_bounds__(256)
void msg_kernel(const float* __restrict__ Qf, const float* __restrict__ KV,
                const float* __restrict__ Ksum, bf16_t* __restrict__ msgb) {
  int blk = blockIdx.x;                         // b(8) * h(8) * lchunk(512)
  int lc = blk & 511, h = (blk >> 9) & 7, b = blk >> 12;
  __shared__ float skv[1024];
  __shared__ float sks[32];
  __shared__ float sq[8][32];
  int t = threadIdx.x;
  for (int i = t; i < 1024; i += 256) skv[i] = KV[((size_t)b * 8 + h) * 1024 + i];
  if (t < 32) sks[t] = Ksum[(b * 8 + h) * 32 + t];
  int wave = t >> 5, lane = t & 31;
  int l = lc * 8 + wave;
  sq[wave][lane] = Qf[((size_t)b * 4096 + l) * 256 + h * 32 + lane];
  __syncthreads();
  float p = sq[wave][lane] * sks[lane];
#pragma unroll
  for (int o = 16; o > 0; o >>= 1) p += __shfl_xor(p, o, 32);
  float z = 1.f / (p + 1e-6f);
  float m = 0.f;
#pragma unroll
  for (int d2 = 0; d2 < 32; ++d2) m += sq[wave][d2] * skv[d2 * 32 + lane];
  msgb[((size_t)b * 4096 + l) * 256 + h * 32 + lane] = f2bf(m * z * 256.f);
}

// ---------------- host orchestration ---------------------------------------
extern "C" void kernel_launch(void* const* d_in, const int* in_sizes, int n_in,
                              void* d_out, int out_size, void* d_ws, size_t ws_size,
                              hipStream_t stream) {
  (void)in_sizes; (void)n_in; (void)out_size; (void)ws_size;
  const float* x      = (const float*)d_in[0];
  const float* sr_w   = (const float*)d_in[1];
  const float* sr_b   = (const float*)d_in[2];
  const float* norm_g = (const float*)d_in[3];
  const float* norm_b = (const float*)d_in[4];
  const float* wq     = (const float*)d_in[5];
  const float* wk     = (const float*)d_in[6];
  const float* wv     = (const float*)d_in[7];
  const float* wm     = (const float*)d_in[8];
  const float* w1     = (const float*)d_in[9];
  const float* w2     = (const float*)d_in[10];
  const float* n1g    = (const float*)d_in[11];
  const float* n1b    = (const float*)d_in[12];
  const float* n2g    = (const float*)d_in[13];
  const float* n2b    = (const float*)d_in[14];
  float* out = (float*)d_out;

  char* ws = (char*)d_ws;
  size_t off = 0;
  auto carve = [&](size_t bytes) -> char* {
    char* p = ws + off;
    off += (bytes + 255) & ~(size_t)255;
    return p;
  };
  bf16_t* cat   = (bf16_t*)carve(32768ULL * 512 * 2);  // [x_bf | msg_ln_bf]
  bf16_t* Aconv = (bf16_t*)carve(2048ULL * 4096 * 2);
  bf16_t* Wc    = (bf16_t*)carve(256ULL * 4096 * 2);   // [N=256, K=4096]
  bf16_t* wqB   = (bf16_t*)carve(256ULL * 256 * 2);    // native [N,K]
  bf16_t* wkB   = (bf16_t*)carve(256ULL * 256 * 2);
  bf16_t* wvB   = (bf16_t*)carve(256ULL * 256 * 2);
  bf16_t* wmB   = (bf16_t*)carve(256ULL * 256 * 2);
  bf16_t* w1B   = (bf16_t*)carve(512ULL * 512 * 2);
  bf16_t* w2B   = (bf16_t*)carve(256ULL * 512 * 2);
  float*  xs    = (float*)carve(2048ULL * 256 * 4);
  bf16_t* srcb  = (bf16_t*)carve(2048ULL * 256 * 2);
  float*  QQ    = (float*)carve(32768ULL * 256 * 4);   // Q, then msg2, then h2
  float*  Kf    = (float*)carve(2048ULL * 256 * 4);
  float*  Vf    = (float*)carve(2048ULL * 256 * 4);
  float*  KVb   = (float*)carve(64ULL * 1024 * 4);
  float*  Ksum  = (float*)carve(64ULL * 32 * 4);
  bf16_t* msgb  = (bf16_t*)carve(32768ULL * 256 * 2);
  bf16_t* h1b   = (bf16_t*)carve(32768ULL * 512 * 2);

  dim3 blk(256);

  // packing
  pack_x_concat<<<32768, blk, 0, stream>>>(x, cat);
  pack_conv_a<<<32768, blk, 0, stream>>>(x, Aconv);
  pack_conv_w<<<4096, blk, 0, stream>>>(sr_w, Wc);
  pack_cvt<<<256, blk, 0, stream>>>(wq, wqB, 65536);
  pack_cvt<<<256, blk, 0, stream>>>(wk, wkB, 65536);
  pack_cvt<<<256, blk, 0, stream>>>(wv, wvB, 65536);
  pack_cvt<<<256, blk, 0, stream>>>(wm, wmB, 65536);
  pack_cvt<<<1024, blk, 0, stream>>>(w1, w1B, 262144);
  pack_cvt<<<512, blk, 0, stream>>>(w2, w2B, 131072);

  // strided-conv subsample as GEMM + bias, then LN -> src bf16
  gemm_bf16_wmma<EPI_BIAS><<<dim3(16, 2), blk, 0, stream>>>(
      Aconv, 4096, Wc, 4096, xs, nullptr, 256, 4096, sr_b);
  ln256<<<2048, blk, 0, stream>>>(xs, norm_g, norm_b, nullptr, srcb, 256, 0, nullptr);

  // projections (Q/K get fused elu+1)
  gemm_bf16_wmma<EPI_ELU1><<<dim3(256, 2), blk, 0, stream>>>(
      cat, 512, wqB, 256, QQ, nullptr, 256, 256, nullptr);
  gemm_bf16_wmma<EPI_ELU1><<<dim3(16, 2), blk, 0, stream>>>(
      srcb, 256, wkB, 256, Kf, nullptr, 256, 256, nullptr);
  gemm_bf16_wmma<EPI_NONE><<<dim3(16, 2), blk, 0, stream>>>(
      srcb, 256, wvB, 256, Vf, nullptr, 256, 256, nullptr);

  // linear attention
  kv_reduce<<<64, blk, 0, stream>>>(Kf, Vf, KVb, Ksum);
  msg_kernel<<<32768, blk, 0, stream>>>(QQ, KVb, Ksum, msgb);

  // merge proj + LN -> right half of concat
  gemm_bf16_wmma<EPI_NONE><<<dim3(256, 2), blk, 0, stream>>>(
      msgb, 256, wmB, 256, QQ, nullptr, 256, 256, nullptr);
  ln256<<<32768, blk, 0, stream>>>(QQ, n1g, n1b, nullptr, cat, 512, 256, nullptr);

  // MLP: relu(concat @ w1^T) @ w2^T
  gemm_bf16_wmma<EPI_RELU><<<dim3(256, 4), blk, 0, stream>>>(
      cat, 512, w1B, 512, nullptr, h1b, 512, 512, nullptr);
  gemm_bf16_wmma<EPI_NONE><<<dim3(256, 2), blk, 0, stream>>>(
      h1b, 512, w2B, 512, QQ, nullptr, 256, 512, nullptr);

  // final LN + residual -> out
  ln256<<<32768, blk, 0, stream>>>(QQ, n2g, n2b, out, nullptr, 0, 0, x);
}
